// MultiHeadAttention_58695023067217
// MI455X (gfx1250) — matrix operand order repaired and checked by reading, not compiled, not run on hardware
//
#include <hip/hip_runtime.h>
#include <hip/hip_bf16.h>
#include <stdint.h>

// ---------------------------------------------------------------------------
// MI455X / gfx1250 multi-head attention (RoPE + causal flash attention)
// - all matmuls on v_wmma_f32_16x16x32_f16 (f32 accumulate), wave32
// - LDS layouts chosen so every WMMA fragment load is ds_load_b128
// - flash Q/K tiles staged with global_load_async_to_lds_b128 (ASYNCcnt)
// ---------------------------------------------------------------------------

typedef __attribute__((ext_vector_type(16))) _Float16 v16h;
typedef __attribute__((ext_vector_type(8)))  _Float16 v8h;
typedef __attribute__((ext_vector_type(8)))  float    v8f;

__device__ __forceinline__ v8f wmma_f16(v16h a, v16h b, v8f c) {
  // D = A(16x32 f16) * B(32x16 f16) + C(16x16 f32)
  return __builtin_amdgcn_wmma_f32_16x16x32_f16(
      /*neg_a=*/false, a, /*neg_b=*/false, b,
      /*c_mod=*/(short)0, c, /*reuse_a=*/false, /*reuse_b=*/false);
}

__device__ __forceinline__ v16h cat16(v8h lo, v8h hi) {
  return __builtin_shufflevector(lo, hi, 0,1,2,3,4,5,6,7,8,9,10,11,12,13,14,15);
}

// A fragment: 16 rows x 32 K f16, LDS row-major [row][k], stride ldk (mult of 8).
// Lane hl=row; this code's element order is K = hi*16..+15 (same as B loads) -> two b128.
__device__ __forceinline__ v16h load_A16(const _Float16* base, int ldk, int lane) {
  const int hl = lane & 15, hi = lane >> 4;
  const _Float16* r = base + hl * ldk + hi * 16;
  return cat16(*(const v8h*)r, *(const v8h*)(r + 8));
}

// B fragment: 32 K x 16 cols f16, from LDS array R[n][k] (row per output col n,
// contiguous K), stride ld (mult of 8).  Lane: n=hl, K=hi*16..+15 -> two b128.
__device__ __forceinline__ v16h load_B16_rows(const _Float16* base, int ld, int lane) {
  const int hl = lane & 15, hi = lane >> 4;
  const _Float16* r = base + hl * ld + hi * 16;
  return cat16(*(const v8h*)r, *(const v8h*)(r + 8));
}

// B fragment from fragment-major staging Bf[lane][16] -> two b128.
__device__ __forceinline__ v16h load_B16_packed(const _Float16* base, int lane) {
  const _Float16* r = base + lane * 16;
  return cat16(*(const v8h*)r, *(const v8h*)(r + 8));
}

// 64B-per-thread async global->LDS copy (4x b128; offset applies to both sides)
__device__ __forceinline__ void async_copy_64B(const _Float16* gsrc, _Float16* ldst) {
  const uint32_t lds = (uint32_t)(uintptr_t)ldst;        // low 32 bits = LDS offset
  const uint64_t gaddr = (uint64_t)(uintptr_t)gsrc;
  asm volatile(
      "global_load_async_to_lds_b128 %0, %1, off\n\t"
      "global_load_async_to_lds_b128 %0, %1, off offset:16\n\t"
      "global_load_async_to_lds_b128 %0, %1, off offset:32\n\t"
      "global_load_async_to_lds_b128 %0, %1, off offset:48"
      :: "v"(lds), "v"(gaddr) : "memory");
}

__device__ __forceinline__ void wait_async0() {
  asm volatile("s_wait_asynccnt 0" ::: "memory");
}

// ---------------------------------------------------------------------------
// Kernel 1: Y = X @ W^T  (einsum 'bsd,ed->bse'), f32 in, f16 out in
// [b,h,s,d_k] layout, optional RoPE in the epilogue.
// Block = 256 thr (8 waves), tile 64(M) x 64(N), K-step 32.
// ---------------------------------------------------------------------------
__global__ __launch_bounds__(256)
void proj_rope_kernel(const float* __restrict__ X, const float* __restrict__ W,
                      const int* __restrict__ pos, _Float16* __restrict__ out,
                      int rope) {
  const int K = 1024, S = 2048, H = 16;
  __shared__ _Float16 As[64][40];      // [m][k] (+pad), A-frag friendly
  __shared__ _Float16 Bf[4][32][16];   // fragment-major: [n>>4][(k>>4)*16+(n&15)][k&15]

  const int r0 = blockIdx.x * 64, n0 = blockIdx.y * 64;
  const int tid = threadIdx.x, lane = tid & 31, wid = tid >> 5;
  const int wm = wid & 3, wn = wid >> 2;     // 4 waves in M, 2 in N

  v8f acc0 = {}, acc1 = {};

  for (int k0 = 0; k0 < K; k0 += 32) {
    // ---- stage A tile (64x32 f32 -> f16), one v8h LDS store per thread
    {
      const int row = tid >> 2, c = (tid & 3) * 8;
      const float4* src = (const float4*)(X + (size_t)(r0 + row) * K + k0 + c);
      float4 f0 = src[0], f1 = src[1];
      v8h h;
      h[0]=(_Float16)f0.x; h[1]=(_Float16)f0.y; h[2]=(_Float16)f0.z; h[3]=(_Float16)f0.w;
      h[4]=(_Float16)f1.x; h[5]=(_Float16)f1.y; h[6]=(_Float16)f1.z; h[7]=(_Float16)f1.w;
      *(v8h*)&As[row][c] = h;
    }
    // ---- stage B tile fragment-major, one v8h LDS store per thread
    {
      const int n = tid >> 2, c = (tid & 3) * 8;   // B[k=c..c+7][n] = W[n0+n][k0+c..]
      const float4* src = (const float4*)(W + (size_t)(n0 + n) * K + k0 + c);
      float4 f0 = src[0], f1 = src[1];
      v8h h;
      h[0]=(_Float16)f0.x; h[1]=(_Float16)f0.y; h[2]=(_Float16)f0.z; h[3]=(_Float16)f0.w;
      h[4]=(_Float16)f1.x; h[5]=(_Float16)f1.y; h[6]=(_Float16)f1.z; h[7]=(_Float16)f1.w;
      *(v8h*)&Bf[n >> 4][((c >> 4) * 16) + (n & 15)][c & 15] = h;
    }
    __syncthreads();

    v16h a  = load_A16(&As[wm*16][0], 40, lane);
    v16h b0 = load_B16_packed(&Bf[wn*2 + 0][0][0], lane);
    v16h b1 = load_B16_packed(&Bf[wn*2 + 1][0][0], lane);
    acc0 = wmma_f16(a, b0, acc0);
    acc1 = wmma_f16(a, b1, acc1);
    __syncthreads();
  }

  // ---- epilogue: optional RoPE, store f16 [b][h][s][dk]
  const int hl = lane & 15, hi = lane >> 4;
#pragma unroll
  for (int j = 0; j < 2; ++j) {
    v8f acc = j ? acc1 : acc0;
    const int col = n0 + wn*32 + j*16 + hl;
    const int h = col >> 6, dk = col & 63;
#pragma unroll
    for (int i = 0; i < 8; ++i) {
      const int grow = r0 + wm*16 + hi*8 + i;   // b*2048 + s
      float v = acc[i];
      if (rope) {
        float p   = (float)pos[grow];
        float ang = p * __powf(10000.0f, -(float)(dk & ~1) * (1.0f/64.0f));
        float cs = __cosf(ang), sn = __sinf(ang);
        float partner = __shfl_xor(v, 1, 32);   // even<->odd dk in adjacent lanes
        v = (dk & 1) ? (sn*partner + cs*v) : (cs*v - sn*partner);
      }
      const int bb = grow >> 11, ss = grow & 2047;
      out[(((size_t)bb*H + h)*S + ss)*64 + dk] = (_Float16)v;
    }
  }
}

// ---------------------------------------------------------------------------
// Kernel 2: causal flash attention.  Block = 128 thr (4 waves), one (b,h) and
// one 64-row Q tile per block; each wave owns 16 query rows.
// S = (Q K^T)/8 with online softmax, O += P V.  Q/K/V f16, stats f32.
// Q and K tiles staged via async global->LDS DMA; V transposed through VGPRs.
// ---------------------------------------------------------------------------
__global__ __launch_bounds__(128)
void flash_attn_kernel(const _Float16* __restrict__ Q,
                       const _Float16* __restrict__ Kt,
                       const _Float16* __restrict__ V,
                       _Float16* __restrict__ O) {
  const int S = 2048, DK = 64, H = 16;
  __shared__ _Float16 Qs[64][72];       // [q][dk]      (A-frags for Q K^T)
  __shared__ _Float16 Ks[64][72];       // [key][dk]    (B-frags for Q K^T, row=n)
  __shared__ _Float16 Vt[64][72];       // [dk][key]    (B-frags for P V, row=n)
  __shared__ _Float16 Ps[4][16][72];    // per-wave P   (A-frags for P V)

  const int q0 = blockIdx.x * 64;
  const int bh = blockIdx.y;            // b*H + h
  const int tid = threadIdx.x, lane = tid & 31, w = tid >> 5;
  const int hl = lane & 15, hi = lane >> 4;

  const _Float16* Qg = Q  + (size_t)bh * S * DK;
  const _Float16* Kg = Kt + (size_t)bh * S * DK;
  const _Float16* Vg = V  + (size_t)bh * S * DK;

  const int r = tid >> 1, c = (tid & 1) * 32;   // per-thread copy slot (64B)

  // ---- Q tile once: async DMA straight into LDS
  async_copy_64B(Qg + (size_t)(q0 + r) * DK + c, &Qs[r][c]);
  wait_async0();
  __syncthreads();
  const v16h qa0 = load_A16(&Qs[w*16][0],  72, lane);
  const v16h qa1 = load_A16(&Qs[w*16][32], 72, lane);

  float run_max[8], run_sum[8];
#pragma unroll
  for (int i = 0; i < 8; ++i) { run_max[i] = -1e30f; run_sum[i] = 0.f; }
  v8f oacc[4] = {v8f{}, v8f{}, v8f{}, v8f{}};

  for (int kt = 0; kt <= (int)blockIdx.x; ++kt) {
    const int kb = kt * 64;
    __syncthreads();                           // WAR on Ks/Vt
    // K tile: async DMA, row-major (fragment rows == key rows)
    async_copy_64B(Kg + (size_t)(kb + r) * DK + c, &Ks[r][c]);
    // V tile: transpose to [dk][key] through VGPRs (overlaps with async K)
    {
      const v8h* vsrc = (const v8h*)(Vg + (size_t)(kb + r) * DK + c);
#pragma unroll
      for (int q8 = 0; q8 < 4; ++q8) {
        v8h t = vsrc[q8];
#pragma unroll
        for (int j = 0; j < 8; ++j) Vt[c + q8*8 + j][r] = t[j];
      }
      if (kt < (int)blockIdx.x)                // prefetch next K tile
        __builtin_prefetch(Kg + (size_t)(kb + 64 + r) * DK + c, 0, 1);
    }
    wait_async0();
    __syncthreads();

    // ---- S = Q K^T (per wave: 16 q-rows x 64 keys, 8 WMMAs)
    v8f sacc[4] = {v8f{}, v8f{}, v8f{}, v8f{}};
#pragma unroll
    for (int j = 0; j < 4; ++j) {
      v16h b0 = load_B16_rows(&Ks[j*16][0],  72, lane);
      v16h b1 = load_B16_rows(&Ks[j*16][32], 72, lane);
      sacc[j] = wmma_f16(qa0, b0, sacc[j]);
      sacc[j] = wmma_f16(qa1, b1, sacc[j]);
    }

    // ---- scale + causal mask (only the diagonal tile needs masking)
#pragma unroll
    for (int j = 0; j < 4; ++j) {
#pragma unroll
      for (int i = 0; i < 8; ++i) {
        float s = sacc[j][i] * 0.125f;         // 1/sqrt(64)
        if (kt == (int)blockIdx.x) {
          const int qpos = q0 + w*16 + hi*8 + i;
          const int kpos = kb + j*16 + hl;
          if (kpos > qpos) s = -1e30f;
        }
        sacc[j][i] = s;
      }
    }

    // ---- online softmax (row stats shared across each 16-lane half-wave)
#pragma unroll
    for (int i = 0; i < 8; ++i) {
      float tm = fmaxf(fmaxf(sacc[0][i], sacc[1][i]), fmaxf(sacc[2][i], sacc[3][i]));
#pragma unroll
      for (int off = 8; off >= 1; off >>= 1) tm = fmaxf(tm, __shfl_xor(tm, off, 32));
      const float nm   = fmaxf(run_max[i], tm);
      const float corr = __expf(run_max[i] - nm);
      run_max[i] = nm;
      float ls = 0.f;
#pragma unroll
      for (int j = 0; j < 4; ++j) {
        float p = __expf(sacc[j][i] - nm);
        sacc[j][i] = p;
        ls += p;
      }
#pragma unroll
      for (int off = 8; off >= 1; off >>= 1) ls += __shfl_xor(ls, off, 32);
      run_sum[i] = run_sum[i] * corr + ls;
#pragma unroll
      for (int nj = 0; nj < 4; ++nj) oacc[nj][i] *= corr;
    }

    // ---- P -> per-wave LDS (C-layout -> A-layout transpose, scalar by nature)
#pragma unroll
    for (int j = 0; j < 4; ++j)
#pragma unroll
      for (int i = 0; i < 8; ++i)
        Ps[w][hi*8 + i][j*16 + hl] = (_Float16)sacc[j][i];
    asm volatile("s_wait_dscnt 0" ::: "memory");  // intra-wave LDS RAW

    // ---- O += P V  (8 WMMAs)
    const v16h pa0 = load_A16(&Ps[w][0][0],  72, lane);
    const v16h pa1 = load_A16(&Ps[w][0][32], 72, lane);
#pragma unroll
    for (int nj = 0; nj < 4; ++nj) {
      v16h vb0 = load_B16_rows(&Vt[nj*16][0],  72, lane);
      v16h vb1 = load_B16_rows(&Vt[nj*16][32], 72, lane);
      oacc[nj] = wmma_f16(pa0, vb0, oacc[nj]);
      oacc[nj] = wmma_f16(pa1, vb1, oacc[nj]);
    }
  }

  // ---- finalize: O /= sum, store f16 [row = b*2048+s][e = h*64+dk]
  const int b = bh >> 4, h = bh & 15;
  float inv[8];
#pragma unroll
  for (int i = 0; i < 8; ++i) inv[i] = 1.0f / run_sum[i];
#pragma unroll
  for (int nj = 0; nj < 4; ++nj)
#pragma unroll
    for (int i = 0; i < 8; ++i) {
      const int row = b*2048 + q0 + w*16 + hi*8 + i;
      const int e   = h*64 + nj*16 + hl;
      O[(size_t)row * 1024 + e] = (_Float16)(oacc[nj][i] * inv[i]);
    }
}

// ---------------------------------------------------------------------------
// Kernel 3: out = attn @ W_O^T (einsum 'bse,de->bsd'), f16 A, f32 out.
// ---------------------------------------------------------------------------
__global__ __launch_bounds__(256)
void outproj_kernel(const _Float16* __restrict__ A, const float* __restrict__ W,
                    float* __restrict__ out) {
  const int K = 1024;
  __shared__ _Float16 As[64][40];
  __shared__ _Float16 Bf[4][32][16];

  const int r0 = blockIdx.x * 64, n0 = blockIdx.y * 64;
  const int tid = threadIdx.x, lane = tid & 31, wid = tid >> 5;
  const int wm = wid & 3, wn = wid >> 2;

  v8f acc0 = {}, acc1 = {};

  for (int k0 = 0; k0 < K; k0 += 32) {
    {
      const int row = tid >> 2, c = (tid & 3) * 8;
      *(v8h*)&As[row][c] = *(const v8h*)(A + (size_t)(r0 + row) * K + k0 + c);
    }
    {
      const int n = tid >> 2, c = (tid & 3) * 8;
      const float4* src = (const float4*)(W + (size_t)(n0 + n) * K + k0 + c);
      float4 f0 = src[0], f1 = src[1];
      v8h h;
      h[0]=(_Float16)f0.x; h[1]=(_Float16)f0.y; h[2]=(_Float16)f0.z; h[3]=(_Float16)f0.w;
      h[4]=(_Float16)f1.x; h[5]=(_Float16)f1.y; h[6]=(_Float16)f1.z; h[7]=(_Float16)f1.w;
      *(v8h*)&Bf[n >> 4][((c >> 4) * 16) + (n & 15)][c & 15] = h;
    }
    __syncthreads();
    v16h a  = load_A16(&As[wm*16][0], 40, lane);
    v16h b0 = load_B16_packed(&Bf[wn*2 + 0][0][0], lane);
    v16h b1 = load_B16_packed(&Bf[wn*2 + 1][0][0], lane);
    acc0 = wmma_f16(a, b0, acc0);
    acc1 = wmma_f16(a, b1, acc1);
    __syncthreads();
  }

  const int hl = lane & 15, hi = lane >> 4;
#pragma unroll
  for (int j = 0; j < 2; ++j) {
    v8f acc = j ? acc1 : acc0;
    const int col = n0 + wn*32 + j*16 + hl;
#pragma unroll
    for (int i = 0; i < 8; ++i) {
      const int grow = r0 + wm*16 + hi*8 + i;
      out[(size_t)grow * 1024 + col] = acc[i];
    }
  }
}

// ---------------------------------------------------------------------------
extern "C" void kernel_launch(void* const* d_in, const int* in_sizes, int n_in,
                              void* d_out, int out_size, void* d_ws, size_t ws_size,
                              hipStream_t stream) {
  (void)in_sizes; (void)n_in; (void)out_size; (void)ws_size;
  const float* x   = (const float*)d_in[0];
  const int*   pos = (const int*)d_in[1];
  const float* WQ  = (const float*)d_in[2];
  const float* WK  = (const float*)d_in[3];
  const float* WV  = (const float*)d_in[4];
  const float* WO  = (const float*)d_in[5];
  float* out = (float*)d_out;

  const int B = 2, S = 2048, D = 1024, H = 16;
  const size_t rows = (size_t)B * S;           // 4096
  const size_t per  = rows * D;                // 4M halves = 8MB each

  _Float16* q    = (_Float16*)d_ws;            // [b,h,s,dk] f16
  _Float16* k    = q + per;
  _Float16* v    = k + per;
  _Float16* attn = v + per;                    // [row,e] f16

  dim3 gblk(256);
  dim3 ggrd(rows / 64, D / 64);                // (64, 16)
  proj_rope_kernel<<<ggrd, gblk, 0, stream>>>(x, WQ, pos, q, 1);
  proj_rope_kernel<<<ggrd, gblk, 0, stream>>>(x, WK, pos, k, 1);
  proj_rope_kernel<<<ggrd, gblk, 0, stream>>>(x, WV, pos, v, 0);

  dim3 agrd(S / 64, B * H);                    // (32, 32)
  flash_attn_kernel<<<agrd, dim3(128), 0, stream>>>(q, k, v, attn);

  outproj_kernel<<<ggrd, gblk, 0, stream>>>(attn, WO, out);
}
